// KernelAttention_56246891708559
// MI455X (gfx1250) — matrix operand order repaired and checked by reading, not compiled, not run on hardware
//
#include <hip/hip_runtime.h>

// ---------------- problem constants ----------------
#define CDIM 256      // channels
#define DDIM 32       // qk head dim
#define NPIX 16384    // H*W
#define TILE_P 128    // pixels per workgroup in k1/k3
#define NTILES 128    // NPIX / TILE_P

typedef unsigned int u32;
typedef unsigned long long u64;
typedef unsigned short u16;
typedef __attribute__((ext_vector_type(4))) u32 u32x4;
typedef __attribute__((ext_vector_type(4))) int i32x4;
typedef __attribute__((ext_vector_type(8))) int i32x8;
typedef __attribute__((ext_vector_type(16))) __bf16 v16bf;
typedef __attribute__((ext_vector_type(8))) float v8f;

// 32-byte WMMA operand fragment (16 bf16 per lane)
union Frag {
  v16bf v;
  u32x4 q[2];
  u32 u[8];
};

__device__ __forceinline__ u16 f2bf(float f) {
  u32 u = __float_as_uint(f);
  u += 0x7fffu + ((u >> 16) & 1u);   // round-to-nearest-even
  return (u16)(u >> 16);
}
__device__ __forceinline__ float bf2f(u16 h) {
  return __uint_as_float(((u32)h) << 16);
}
__device__ __forceinline__ u32 pack2(float a, float b) {
  return (u32)f2bf(a) | ((u32)f2bf(b) << 16);
}
__device__ __forceinline__ float softplus_f(float x) {
  return (x > 20.0f) ? x : log1pf(__expf(x));
}

__device__ __forceinline__ v8f wmma_bf16(const Frag& a, const Frag& b, v8f c) {
  return __builtin_amdgcn_wmma_f32_16x16x32_bf16(false, a.v, false, b.v,
                                                 (short)0, c, false, false);
}

// A fragment (16x32 bf16): lane holds row = lane&15, 32 contiguous bytes at half*32.
__device__ __forceinline__ void load_fragA(Frag& f, const char* base, int byteOff) {
  f.q[0] = *(const u32x4*)(base + byteOff);
  f.q[1] = *(const u32x4*)(base + byteOff + 16);
}
// B fragment (32x16 bf16): lane holds column = lane&15, 32 contiguous bytes.
__device__ __forceinline__ void load_fragB(Frag& f, const char* base, int byteOff) {
  f.q[0] = *(const u32x4*)(base + byteOff);
  f.q[1] = *(const u32x4*)(base + byteOff + 16);
}

// Issue one TDM 1-D copy: nelem bf16 elements global -> LDS (wave-level op).
__device__ __forceinline__ void tdm_copy_b16(u32 lds_off, const void* gptr,
                                             u32 nelem) {
  u64 ga = (u64)(uintptr_t)gptr;
  u32x4 g0;
  g0[0] = 1u;                                   // count=1, user descriptor
  g0[1] = lds_off;                              // lds_addr
  g0[2] = (u32)ga;                              // global_addr[31:0]
  g0[3] = (u32)((ga >> 32) & 0x01FFFFFFull) | (2u << 30);  // addr[56:32], type=2
  i32x8 g1;
  g1[0] = (int)(1u << 16);                      // data_size=1 (2 bytes)
  g1[1] = (int)(nelem << 16);                   // tensor_dim0 lo16
  g1[2] = (int)((nelem >> 16) & 0xffffu) | (1 << 16);  // dim0 hi16, tensor_dim1=1
  g1[3] = (int)(nelem << 16);                   // tile_dim0 = nelem
  g1[4] = 1;                                    // tile_dim1 = 1
  g1[5] = (int)nelem;                           // tensor_dim0_stride
  g1[6] = 0;
  g1[7] = 0;
  i32x4 g2 = {0, 0, 0, 0};
  i32x4 g3 = {0, 0, 0, 0};
#if defined(__clang_major__) && (__clang_major__ >= 23)
  i32x8 g4 = {0, 0, 0, 0, 0, 0, 0, 0};
  __builtin_amdgcn_tensor_load_to_lds(g0, g1, g2, g3, g4, 0);
#else
  __builtin_amdgcn_tensor_load_to_lds(g0, g1, g2, g3, 0);
#endif
}

// ---------------- small prep kernels ----------------
__global__ void k_zero(float* __restrict__ p, int n) {
  int i = blockIdx.x * blockDim.x + threadIdx.x;
  if (i < n) p[i] = 0.0f;
}
__global__ void k_f2bf(const float* __restrict__ a, u16* __restrict__ o, int n) {
  int i = blockIdx.x * blockDim.x + threadIdx.x;
  if (i < n) o[i] = f2bf(a[i]);
}

// ---------------- pass 1: Q/K projections + Ksum + KX ----------------
__global__ void __launch_bounds__(256)
k1_proj(const float* __restrict__ x,
        const u16* __restrict__ Wq, const u16* __restrict__ Wk,
        const float* __restrict__ bq, const float* __restrict__ bk,
        u16* __restrict__ Qout, float* __restrict__ KX,
        float* __restrict__ Ksum) {
  extern __shared__ char smem[];
  u16* Xpc  = (u16*)(smem);            // [128 pix][256 ch] bf16
  u16* Xcp  = (u16*)(smem + 65536);    // [256 ch][128 pix] bf16
  u16* Ktms = (u16*)(smem + 131072);   // [32][128] softplus(K)
  u16* Qpm  = (u16*)(smem + 139264);   // [128][32] softplus(Q)

  const int tid = threadIdx.x;
  const int b = blockIdx.x >> 7;
  const int n0 = (blockIdx.x & (NTILES - 1)) * TILE_P;
  const float* xb = x + (size_t)b * CDIM * NPIX;

  for (int i = 0; i < 32; ++i) {
    int lin = i * 256 + tid;
    int c = lin >> 5;
    int p = (lin & 31) << 2;
    const float* src = xb + (size_t)c * NPIX + n0 + p;
    if (i + 1 < 32) __builtin_prefetch(src + 8 * NPIX, 0, 1);
    float4 v = *(const float4*)src;
    u16 h0 = f2bf(v.x), h1 = f2bf(v.y), h2 = f2bf(v.z), h3 = f2bf(v.w);
    *(u32*)(Xcp + c * TILE_P + p)     = (u32)h0 | ((u32)h1 << 16);
    *(u32*)(Xcp + c * TILE_P + p + 2) = (u32)h2 | ((u32)h3 << 16);
    Xpc[(p + 0) * CDIM + c] = h0;
    Xpc[(p + 1) * CDIM + c] = h1;
    Xpc[(p + 2) * CDIM + c] = h2;
    Xpc[(p + 3) * CDIM + c] = h3;
  }
  __syncthreads();

  const int wave = tid >> 5, lane = tid & 31;
  const int lrow = lane & 15, half = lane >> 4;

  v8f qacc0 = {}, qacc1 = {}, kacc0 = {}, kacc1 = {};
  const char* arow = (const char*)(Xpc + (wave * 16 + lrow) * CDIM);
  const char* wq0 = (const char*)(Wq + lrow * CDIM);
  const char* wq1 = (const char*)(Wq + (16 + lrow) * CDIM);
  const char* wk0 = (const char*)(Wk + lrow * CDIM);
  const char* wk1 = (const char*)(Wk + (16 + lrow) * CDIM);
  for (int kk = 0; kk < 8; ++kk) {
    Frag a, f0, f1;
    load_fragA(a, arow, kk * 64 + half * 32);
    load_fragB(f0, wq0, kk * 64 + half * 32);
    load_fragB(f1, wq1, kk * 64 + half * 32);
    qacc0 = wmma_bf16(a, f0, qacc0);
    qacc1 = wmma_bf16(a, f1, qacc1);
    load_fragB(f0, wk0, kk * 64 + half * 32);
    load_fragB(f1, wk1, kk * 64 + half * 32);
    kacc0 = wmma_bf16(a, f0, kacc0);
    kacc1 = wmma_bf16(a, f1, kacc1);
  }
  {
    int m0 = lrow, m1 = 16 + lrow;
    float bq0 = bq[m0], bq1 = bq[m1], bk0 = bk[m0], bk1 = bk[m1];
    int pbase = wave * 16 + half * 8;
    for (int r = 0; r < 8; ++r) {
      int p = pbase + r;
      Qpm[p * DDIM + m0]    = f2bf(softplus_f(qacc0[r] + bq0));
      Qpm[p * DDIM + m1]    = f2bf(softplus_f(qacc1[r] + bq1));
      Ktms[m0 * TILE_P + p] = f2bf(softplus_f(kacc0[r] + bk0));
      Ktms[m1 * TILE_P + p] = f2bf(softplus_f(kacc1[r] + bk1));
    }
  }
  __syncthreads();

  if (tid < DDIM) {
    float s = 0.0f;
    for (int p = 0; p < TILE_P; ++p) s += bf2f(Ktms[tid * TILE_P + p]);
    atomicAdd(&Ksum[b * DDIM + tid], s);
  }

  {
    u32* qg = (u32*)(Qout + ((size_t)b * NPIX + n0) * DDIM);
    const u32* qs = (const u32*)Qpm;
    for (int i = 0; i < 8; ++i) qg[i * 256 + tid] = qs[i * 256 + tid];
  }

  const int mt = wave & 1;
  const int ct4 = wave >> 1;
  const char* aK = (const char*)(Ktms + (mt * 16 + lrow) * TILE_P);
  for (int j = 0; j < 4; ++j) {
    int ctile = ct4 * 4 + j;
    const char* bX = (const char*)(Xcp + (ctile * 16 + lrow) * TILE_P);
    v8f acc = {};
    for (int ks = 0; ks < 4; ++ks) {
      Frag a, f;
      load_fragA(a, aK, ks * 64 + half * 32);
      load_fragB(f, bX, ks * 64 + half * 32);
      acc = wmma_bf16(a, f, acc);
    }
    float* kxb = KX + ((size_t)b * DDIM + mt * 16 + half * 8) * CDIM +
                 ctile * 16 + lrow;
    for (int r = 0; r < 8; ++r) atomicAdd(kxb + (size_t)r * CDIM, acc[r]);
  }
}

// ---------------- pass 2: KVT[b][c][m] = Wv[c,:]·KX[m,:] + bv[c]*Ksum[m] ----
__global__ void __launch_bounds__(256)
k2_kvt(const u16* __restrict__ Wv, const float* __restrict__ bv,
       const float* __restrict__ KX, const float* __restrict__ Ksum,
       u16* __restrict__ KVT) {
  const int tid = threadIdx.x;
  const int wave = tid >> 5, lane = tid & 31;
  const int lrow = lane & 15, half = lane >> 4;
  const int tileId = blockIdx.x * 8 + wave;
  const int b = tileId >> 5;
  const int t = tileId & 31;
  const int ctile = t >> 1, mtile = t & 1;
  const int m = mtile * 16 + lrow;

  const char* aW = (const char*)(Wv + (ctile * 16 + lrow) * CDIM);
  const float* kxr = KX + ((size_t)b * DDIM + m) * CDIM;
  v8f acc = {};
  for (int kk = 0; kk < 8; ++kk) {
    Frag a, f;
    load_fragA(a, aW, kk * 64 + half * 32);
    const float* src = kxr + kk * 32 + half * 16;
    for (int i = 0; i < 8; ++i) f.u[i] = pack2(src[2 * i], src[2 * i + 1]);
    acc = wmma_bf16(a, f, acc);
  }
  float ks = Ksum[b * DDIM + m];
  for (int r = 0; r < 8; ++r) {
    int c = ctile * 16 + half * 8 + r;
    KVT[((size_t)b * CDIM + c) * DDIM + m] = f2bf(acc[r] + bv[c] * ks);
  }
}

// ---------------- pass 3: out = x + gamma * norm * (KVT · Q^T) ----------
// KVT[b] staged via TDM tensor_load_to_lds; Q block via async-to-LDS copies.
__global__ void __launch_bounds__(256)
k3_out(const float* __restrict__ x, const u16* __restrict__ Qbf,
       const u16* __restrict__ KVT, const float* __restrict__ Ksum,
       const float* __restrict__ gamma, float* __restrict__ out) {
  __shared__ __attribute__((aligned(16))) u16 sh_kvt[CDIM * DDIM];  // 16 KB
  __shared__ __attribute__((aligned(16))) u16 sh_q[TILE_P * DDIM];  //  8 KB
  __shared__ float KsumEps[DDIM];

  const int tid = threadIdx.x;
  const int b = blockIdx.x >> 7;
  const int n0 = (blockIdx.x & (NTILES - 1)) * TILE_P;

  // TDM: KVT[b] (8192 bf16) -> LDS, issued once by wave 0.
  if (tid < 32) {
    tdm_copy_b16((u32)(uintptr_t)(void*)sh_kvt,
                 KVT + (size_t)b * CDIM * DDIM, CDIM * DDIM);
    __builtin_amdgcn_s_wait_tensorcnt(0);
  }

  // Async copy: this tile's Q rows (128 x 32 bf16 = 8 KB) -> LDS.
  {
    u32 ldsq = (u32)(uintptr_t)(void*)sh_q + (u32)tid * 16u;
    u64 gq = (u64)(uintptr_t)(Qbf + ((size_t)b * NPIX + n0) * DDIM) +
             (u64)tid * 16u;
    asm volatile("global_load_async_to_lds_b128 %0, %1, off"
                 :: "v"(ldsq), "v"(gq) : "memory");
    asm volatile("global_load_async_to_lds_b128 %0, %1, off offset:4096"
                 :: "v"(ldsq), "v"(gq) : "memory");
    asm volatile("s_wait_asynccnt 0x0" ::: "memory");
  }

  if (tid < DDIM) KsumEps[tid] = Ksum[b * DDIM + tid] + 1e-6f;
  __syncthreads();

  const int wave = tid >> 5, lane = tid & 31;
  const int lrow = lane & 15, half = lane >> 4;
  const int n = n0 + wave * 16 + lrow;   // this lane's pixel (D column)

  // lane's Q row from LDS: 32 bf16 = 64B (norm + B fragment)
  union { u32x4 v[4]; u32 w[16]; } qb;
  {
    const u32x4* qr = (const u32x4*)(sh_q + (wave * 16 + lrow) * DDIM);
    qb.v[0] = qr[0]; qb.v[1] = qr[1]; qb.v[2] = qr[2]; qb.v[3] = qr[3];
  }
  float dot = 0.0f;
  for (int i = 0; i < 16; ++i) {
    u32 w = qb.w[i];
    dot += bf2f((u16)(w & 0xffffu)) * KsumEps[2 * i];
    dot += bf2f((u16)(w >> 16)) * KsumEps[2 * i + 1];
  }
  const float scale = gamma[0] / dot;  // gamma * norm[n]

  Frag qf;
  for (int i = 0; i < 8; ++i) qf.u[i] = qb.w[half * 8 + i];

  const float* xb = x + (size_t)b * CDIM * NPIX;
  float* ob = out + (size_t)b * CDIM * NPIX;
  for (int ct = 0; ct < 16; ++ct) {
    Frag a;
    const char* arow = (const char*)(sh_kvt + (ct * 16 + lrow) * DDIM);
    load_fragA(a, arow, half * 32);
    v8f acc = {};
    acc = wmma_bf16(a, qf, acc);
    int cbase = ct * 16 + half * 8;
    for (int r = 0; r < 8; ++r) {
      size_t idx = (size_t)(cbase + r) * NPIX + n;
      ob[idx] = xb[idx] + scale * acc[r];
    }
  }
}

// ---------------- host launcher ----------------
extern "C" void kernel_launch(void* const* d_in, const int* in_sizes, int n_in,
                              void* d_out, int out_size, void* d_ws,
                              size_t ws_size, hipStream_t stream) {
  const float* x     = (const float*)d_in[0];
  const float* Wq    = (const float*)d_in[1];
  const float* bq    = (const float*)d_in[2];
  const float* Wk    = (const float*)d_in[3];
  const float* bk    = (const float*)d_in[4];
  const float* Wv    = (const float*)d_in[5];
  const float* bv    = (const float*)d_in[6];
  const float* gamma = (const float*)d_in[7];
  float* out = (float*)d_out;
  const int Bn = in_sizes[0] / (CDIM * NPIX);

  char* ws = (char*)d_ws;
  u16* Wq_bf  = (u16*)(ws + 0);        //  16 KB
  u16* Wk_bf  = (u16*)(ws + 16384);    //  16 KB
  u16* Wv_bf  = (u16*)(ws + 32768);    // 128 KB
  u16* KVT    = (u16*)(ws + 163840);   // 128 KB (B*256*32 bf16)
  float* Ksum = (float*)(ws + 294912); //   1 KB (B*32 f32)
  float* KX   = (float*)(ws + 295936); // 256 KB (B*32*256 f32)
  u16* Qbf    = (u16*)(ws + 558080);   //   8 MB (B*N*32 bf16)

  k_f2bf<<<(DDIM * CDIM + 255) / 256, 256, 0, stream>>>(Wq, Wq_bf, DDIM * CDIM);
  k_f2bf<<<(DDIM * CDIM + 255) / 256, 256, 0, stream>>>(Wk, Wk_bf, DDIM * CDIM);
  k_f2bf<<<(CDIM * CDIM + 255) / 256, 256, 0, stream>>>(Wv, Wv_bf, CDIM * CDIM);

  const int nz = Bn * DDIM + Bn * DDIM * CDIM;  // Ksum + KX (contiguous)
  k_zero<<<(nz + 255) / 256, 256, 0, stream>>>(Ksum, nz);

  const unsigned smem1 = 147456;  // 144 KB of the 320 KB WGP LDS pool
  hipFuncSetAttribute((const void*)k1_proj,
                      hipFuncAttributeMaxDynamicSharedMemorySize, (int)smem1);
  k1_proj<<<dim3(Bn * NTILES), dim3(256), smem1, stream>>>(
      x, Wq_bf, Wk_bf, bq, bk, Qbf, KX, Ksum);

  k2_kvt<<<dim3(32), dim3(256), 0, stream>>>(Wv_bf, bv, KX, Ksum, KVT);

  k3_out<<<dim3(Bn * NTILES), dim3(256), 0, stream>>>(x, Qbf, KVT, Ksum, gamma,
                                                      out);
  (void)n_in; (void)out_size; (void)ws_size;
}
